// LlamaAttention_5918464934438
// MI455X (gfx1250) — compile-verified, matrix operand order checked by reading, not run on hardware
//
#include <hip/hip_runtime.h>
#include <hip/hip_bf16.h>

// ---------------------------------------------------------------------------
// LlamaAttention for MI455X (gfx1250): all matmuls via v_wmma_f32_16x16x32_bf16,
// GEMM tile staging via GLOBAL_LOAD_ASYNC_TO_LDS_B128 (double-buffered, ASYNCcnt).
// T=2048, HID=4096, HQ=32, HKV=8, D=128
// ---------------------------------------------------------------------------

#define T_LEN 2048
#define HID   4096
#define HQ    32
#define HKV   8
#define HD    128

typedef __attribute__((ext_vector_type(16))) __bf16 v16bf;
typedef __attribute__((ext_vector_type(8)))  float  v8f;

union Frag16 {
    v16bf v;
    unsigned u[8];
};

static __device__ __forceinline__ unsigned short f2bf(float f) {
    unsigned u = __builtin_bit_cast(unsigned, f);
    u += 0x7FFFu + ((u >> 16) & 1u);   // round-to-nearest-even
    return (unsigned short)(u >> 16);
}

// Per-lane fragment K-offset for the 16-bit A/B layout:
// element pair p (p=0..7, covering elements 2p,2p+1):
//   K = (p>=4 ? 16 : 0) + half*8 + (p&3)*2
static __device__ __forceinline__ int frag_koff(int p, int half) {
    return ((p >> 2) << 4) + (half << 3) + ((p & 3) << 1);
}

// CDNA5 async copy: global -> LDS, 16 bytes per lane, tracked by ASYNCcnt.
static __device__ __forceinline__ void async_copy_b128(unsigned lds_addr,
                                                       unsigned long long gaddr) {
    asm volatile("global_load_async_to_lds_b128 %0, %1, off"
                 :: "v"(lds_addr), "v"(gaddr)
                 : "memory");
}
static __device__ __forceinline__ void wait_async_le5() {
    asm volatile("s_wait_asynccnt 0x5" ::: "memory");
}
static __device__ __forceinline__ void wait_async_le0() {
    asm volatile("s_wait_asynccnt 0x0" ::: "memory");
}

// ---------------------------------------------------------------------------
// Elementwise converts
// ---------------------------------------------------------------------------
__global__ __launch_bounds__(256) void k_f32_to_bf16(const float* __restrict__ src,
                                                     unsigned short* __restrict__ dst,
                                                     int n) {
    int i = blockIdx.x * 256 + threadIdx.x;
    if (i < n) dst[i] = f2bf(src[i]);
}

// src: R x C (row-major f32)  ->  dst: C x R (row-major bf16)
__global__ __launch_bounds__(256) void k_transpose_bf16(const float* __restrict__ src,
                                                        unsigned short* __restrict__ dst,
                                                        int R, int C) {
    int i = blockIdx.x * 256 + threadIdx.x;
    if (i >= R * C) return;
    int r = i / C, c = i % C;
    dst[(size_t)c * R + r] = f2bf(src[(size_t)r * C + c]);
}

// ---------------------------------------------------------------------------
// RoPE: x (T x nheads*128 f32) -> out (bf16), rotate_half convention
// ---------------------------------------------------------------------------
__global__ __launch_bounds__(256) void k_rope_bf16(const float* __restrict__ x,
                                                   const float* __restrict__ cosb,
                                                   const float* __restrict__ sinb,
                                                   unsigned short* __restrict__ out,
                                                   int nheads) {
    int idx = blockIdx.x * 256 + threadIdx.x;
    int total = T_LEN * nheads * HD;
    if (idx >= total) return;
    int d  = idx & (HD - 1);
    int th = idx >> 7;
    int t  = th / nheads;
    float c = cosb[t * HD + d];
    float s = sinb[t * HD + d];
    const float* row = x + (size_t)(idx - d);   // start of this (t,h) vector
    float v   = row[d];
    float rot = (d < 64) ? -row[d + 64] : row[d - 64];
    out[idx] = f2bf(v * c + rot * s);
}

// ---------------------------------------------------------------------------
// NT GEMM: C[MxN] (f32) = A[MxK] (bf16 rowmajor) * B[NxK]^T (bf16 rowmajor)
// Block tile 64x256, 8 waves (2x4), wave tile 32x64 = 2x4 WMMA tiles.
// Double-buffered LDS staged with async global->LDS copies (ASYNCcnt).
// ---------------------------------------------------------------------------
#define GM 64
#define GN 256
#define GK 32

__global__ __launch_bounds__(256) void k_gemm_nt_bf16(const unsigned short* __restrict__ A,
                                                      const unsigned short* __restrict__ B,
                                                      float* __restrict__ C,
                                                      int M, int N, int K) {
    __shared__ unsigned short As[2][GM][GK];   // 2 x 4 KB
    __shared__ unsigned short Bs[2][GN][GK];   // 2 x 16 KB

    int tid  = threadIdx.x;
    int lane = tid & 31;
    int wave = tid >> 5;
    int half = lane >> 4;
    int l15  = lane & 15;

    int bm = blockIdx.y * GM;
    int bn = blockIdx.x * GN;
    int wm = (wave >> 2) * 32;   // 0 or 32
    int wn = (wave & 3) * 64;    // 0,64,128,192

    // per-thread async-copy assignments (16B chunks)
    int rA = tid >> 2,  cA = (tid & 3) * 8;            // A: 64 rows x 4 chunks
    const unsigned long long baseA = (unsigned long long)(uintptr_t)A;
    const unsigned long long baseB = (unsigned long long)(uintptr_t)B;

    v8f acc[2][4] = {};

    int nstages = K / GK;

    // prologue: stage 0 into buffer 0
    {
        async_copy_b128((unsigned)(uintptr_t)&As[0][rA][cA],
                        baseA + 2ull * ((size_t)(bm + rA) * K + cA));
        #pragma unroll
        for (int q = 0; q < 4; q++) {
            int j = tid + q * 256;
            int r = j >> 2, c = (j & 3) * 8;
            async_copy_b128((unsigned)(uintptr_t)&Bs[0][r][c],
                            baseB + 2ull * ((size_t)(bn + r) * K + c));
        }
    }

    for (int s = 0; s < nstages; s++) {
        int buf = s & 1;
        if (s + 1 < nstages) {
            int k1 = (s + 1) * GK;
            async_copy_b128((unsigned)(uintptr_t)&As[buf ^ 1][rA][cA],
                            baseA + 2ull * ((size_t)(bm + rA) * K + k1 + cA));
            #pragma unroll
            for (int q = 0; q < 4; q++) {
                int j = tid + q * 256;
                int r = j >> 2, c = (j & 3) * 8;
                async_copy_b128((unsigned)(uintptr_t)&Bs[buf ^ 1][r][c],
                                baseB + 2ull * ((size_t)(bn + r) * K + k1 + c));
            }
            wait_async_le5();   // newest 5 outstanding -> stage s's copies done
        } else {
            wait_async_le0();
        }
        __syncthreads();

        #pragma unroll
        for (int mi = 0; mi < 2; mi++) {
            Frag16 a;
            int m = wm + mi * 16 + l15;
            #pragma unroll
            for (int p = 0; p < 8; p++)
                a.u[p] = *(const unsigned*)&As[buf][m][frag_koff(p, half)];
            #pragma unroll
            for (int ni = 0; ni < 4; ni++) {
                Frag16 b;
                int n = wn + ni * 16 + l15;
                #pragma unroll
                for (int p = 0; p < 8; p++)
                    b.u[p] = *(const unsigned*)&Bs[buf][n][frag_koff(p, half)];
                acc[mi][ni] = __builtin_amdgcn_wmma_f32_16x16x32_bf16(
                    false, a.v, false, b.v, (short)0, acc[mi][ni], false, false);
            }
        }
        __syncthreads();   // protect buf (rewritten next iteration) until all waves done
    }

    #pragma unroll
    for (int mi = 0; mi < 2; mi++)
        #pragma unroll
        for (int ni = 0; ni < 4; ni++) {
            float* crow = C + (size_t)(bm + wm + mi * 16 + half * 8) * N
                            + (bn + wn + ni * 16 + l15);
            #pragma unroll
            for (int r = 0; r < 8; r++)
                crow[(size_t)r * N] = acc[mi][ni][r];
        }
}

// ---------------------------------------------------------------------------
// Flash attention (causal, GQA 4:1). Q block = 32 rows, K/V block = 64 rows.
// grid = (T/32, HQ), 256 threads (8 waves).
// S (32x64) = 2x4 sixteen-tiles -> one per wave.
// O (32x128) = 2x8 sixteen-tiles -> two per wave.
// ---------------------------------------------------------------------------
#define BR 32
#define BC 64

__global__ __launch_bounds__(256) void k_flash_attn(const unsigned short* __restrict__ Qb,
                                                    const unsigned short* __restrict__ Kb,
                                                    const unsigned short* __restrict__ Vb,
                                                    float* __restrict__ O) {
    __shared__ unsigned short Qs[BR][HD];    // 8 KB
    __shared__ unsigned short Ks[BC][HD];    // 16 KB
    __shared__ unsigned short Vt[HD][BC];    // 16 KB (transposed: [d][key])
    __shared__ float          Ss[BR][BC];    // 8 KB
    __shared__ unsigned short Ps[BR][BC];    // 4 KB
    __shared__ float row_m[BR], row_l[BR], row_a[BR];

    const float scale = 0.08838834764831845f;  // 1/sqrt(128)

    int h   = blockIdx.y;
    int q0  = blockIdx.x * BR;
    int kvh = h >> 2;                 // GQA: 4 query heads per kv head

    int tid  = threadIdx.x;
    int lane = tid & 31;
    int wave = tid >> 5;
    int half = lane >> 4;
    int l15  = lane & 15;
    int smi  = wave >> 2;             // 0..1 : 16-row band of S / O
    int sni  = wave & 3;              // 0..3 : 16-col band of S, 32-col band of O

    // load Q block (rope'd bf16)
    for (int i = tid; i < BR * (HD / 2); i += 256) {
        int r = i / (HD / 2), c = (i % (HD / 2)) * 2;
        *(unsigned*)&Qs[r][c] =
            *(const unsigned*)&Qb[((size_t)(q0 + r) * HQ + h) * HD + c];
    }
    if (tid < BR) { row_m[tid] = -1e30f; row_l[tid] = 0.0f; }

    v8f oacc[2] = {};
    __syncthreads();

    int jend = q0 / BC;   // last key block intersecting the causal triangle
    for (int j = 0; j <= jend; j++) {
        int k0 = j * BC;
        // load K block + V block (V transposed into Vt[d][key])
        for (int i = tid; i < BC * (HD / 2); i += 256) {
            int r = i / (HD / 2), c = (i % (HD / 2)) * 2;
            *(unsigned*)&Ks[r][c] =
                *(const unsigned*)&Kb[((size_t)(k0 + r) * HKV + kvh) * HD + c];
            unsigned uv = *(const unsigned*)&Vb[((size_t)(k0 + r) * HKV + kvh) * HD + c];
            Vt[c][r]     = (unsigned short)(uv & 0xFFFFu);
            Vt[c + 1][r] = (unsigned short)(uv >> 16);
        }
        __syncthreads();

        // ---- S = scale * Q K^T, causal-masked ----
        {
            v8f s = {};
            int m = smi * 16 + l15;
            int n = sni * 16 + l15;
            #pragma unroll
            for (int kk = 0; kk < HD; kk += 32) {
                Frag16 a, b;
                #pragma unroll
                for (int p = 0; p < 8; p++) {
                    int kb = kk + frag_koff(p, half);
                    a.u[p] = *(const unsigned*)&Qs[m][kb];
                    b.u[p] = *(const unsigned*)&Ks[n][kb];
                }
                s = __builtin_amdgcn_wmma_f32_16x16x32_bf16(
                    false, a.v, false, b.v, (short)0, s, false, false);
            }
            #pragma unroll
            for (int r = 0; r < 8; r++) {
                int mm = smi * 16 + r + half * 8;
                int nn = sni * 16 + l15;
                float val = s[r] * scale;
                if (k0 + nn > q0 + mm) val = -1e30f;   // causal mask
                Ss[mm][nn] = val;
            }
        }
        __syncthreads();

        // ---- online softmax (one thread per query row) ----
        if (tid < BR) {
            int r = tid;
            float mold = row_m[r];
            float rmax = -1e30f;
            for (int c = 0; c < BC; c++) rmax = fmaxf(rmax, Ss[r][c]);
            float mnew  = fmaxf(mold, rmax);
            float alpha = __expf(mold - mnew);
            float lsum  = 0.0f;
            for (int c = 0; c < BC; c++) {
                float p = __expf(Ss[r][c] - mnew);
                lsum += p;
                Ps[r][c] = f2bf(p);
            }
            row_l[r] = row_l[r] * alpha + lsum;
            row_m[r] = mnew;
            row_a[r] = alpha;
        }
        __syncthreads();

        // ---- rescale O, then O += P * V ----
        #pragma unroll
        for (int t2 = 0; t2 < 2; t2++)
            #pragma unroll
            for (int r = 0; r < 8; r++)
                oacc[t2][r] *= row_a[smi * 16 + r + half * 8];

        {
            int m = smi * 16 + l15;
            #pragma unroll
            for (int t2 = 0; t2 < 2; t2++) {
                int d0 = sni * 32 + t2 * 16;
                #pragma unroll
                for (int kk = 0; kk < BC; kk += 32) {
                    Frag16 a, b;
                    #pragma unroll
                    for (int p = 0; p < 8; p++) {
                        int kb = kk + frag_koff(p, half);
                        a.u[p] = *(const unsigned*)&Ps[m][kb];
                        b.u[p] = *(const unsigned*)&Vt[d0 + l15][kb];
                    }
                    oacc[t2] = __builtin_amdgcn_wmma_f32_16x16x32_bf16(
                        false, a.v, false, b.v, (short)0, oacc[t2], false, false);
                }
            }
        }
        __syncthreads();
    }

    // ---- normalize and store O (f32, layout T x HQ*128) ----
    #pragma unroll
    for (int t2 = 0; t2 < 2; t2++)
        #pragma unroll
        for (int r = 0; r < 8; r++) {
            int mm = smi * 16 + r + half * 8;
            int dd = sni * 32 + t2 * 16 + l15;
            O[((size_t)(q0 + mm) * HQ + h) * HD + dd] = oacc[t2][r] / row_l[mm];
        }
}

// ---------------------------------------------------------------------------
// Host-side orchestration
// ---------------------------------------------------------------------------
static inline size_t align256(size_t x) { return (x + 255) & ~(size_t)255; }

extern "C" void kernel_launch(void* const* d_in, const int* in_sizes, int n_in,
                              void* d_out, int out_size, void* d_ws, size_t ws_size,
                              hipStream_t stream) {
    (void)in_sizes; (void)n_in; (void)out_size; (void)ws_size;

    const float* Xf   = (const float*)d_in[0];   // T x HID
    // d_in[1] = position_ids (unused; cos/sin are precomputed)
    const float* cosb = (const float*)d_in[2];   // T x 128
    const float* sinb = (const float*)d_in[3];   // T x 128
    const float* Wq   = (const float*)d_in[4];   // HID x HQ*128
    const float* Wk   = (const float*)d_in[5];   // HID x HKV*128
    const float* Wv   = (const float*)d_in[6];   // HID x HKV*128
    const float* Wo   = (const float*)d_in[7];   // HQ*128 x HID
    float* out = (float*)d_out;                  // T x HID

    const int NQ = HQ * HD;     // 4096
    const int NK = HKV * HD;    // 1024

    // workspace carve-up (bf16 stored as unsigned short)
    char* p = (char*)d_ws;
    unsigned short* Xb  = (unsigned short*)p; p += align256((size_t)T_LEN * HID * 2);
    unsigned short* WqT = (unsigned short*)p; p += align256((size_t)NQ * HID * 2);
    unsigned short* WkT = (unsigned short*)p; p += align256((size_t)NK * HID * 2);
    unsigned short* WvT = (unsigned short*)p; p += align256((size_t)NK * HID * 2);
    unsigned short* WoT = (unsigned short*)p; p += align256((size_t)HID * NQ * 2);
    float* Qf = (float*)p; p += align256((size_t)T_LEN * NQ * 4);
    float* Kf = (float*)p; p += align256((size_t)T_LEN * NK * 4);
    float* Vf = (float*)p; p += align256((size_t)T_LEN * NK * 4);
    unsigned short* Qb = (unsigned short*)p; p += align256((size_t)T_LEN * NQ * 2);
    unsigned short* Kb = (unsigned short*)p; p += align256((size_t)T_LEN * NK * 2);
    unsigned short* Vb = (unsigned short*)p; p += align256((size_t)T_LEN * NK * 2);
    float*          attnF = Qf;   // reuse: Qf dead after RoPE->Qb
    unsigned short* attnB = Xb;   // reuse: Xb dead after QKV GEMMs

    // 1) converts / transposes
    {
        int n = T_LEN * HID;
        k_f32_to_bf16<<<(n + 255) / 256, 256, 0, stream>>>(Xf, Xb, n);
    }
    {
        int n = HID * NQ;
        k_transpose_bf16<<<(n + 255) / 256, 256, 0, stream>>>(Wq, WqT, HID, NQ);
        k_transpose_bf16<<<(n + 255) / 256, 256, 0, stream>>>(Wo, WoT, NQ, HID);
    }
    {
        int n = HID * NK;
        k_transpose_bf16<<<(n + 255) / 256, 256, 0, stream>>>(Wk, WkT, HID, NK);
        k_transpose_bf16<<<(n + 255) / 256, 256, 0, stream>>>(Wv, WvT, HID, NK);
    }

    // 2) QKV projections (WMMA NT GEMMs, async double-buffered staging)
    k_gemm_nt_bf16<<<dim3(NQ / GN, T_LEN / GM), 256, 0, stream>>>(Xb, WqT, Qf, T_LEN, NQ, HID);
    k_gemm_nt_bf16<<<dim3(NK / GN, T_LEN / GM), 256, 0, stream>>>(Xb, WkT, Kf, T_LEN, NK, HID);
    k_gemm_nt_bf16<<<dim3(NK / GN, T_LEN / GM), 256, 0, stream>>>(Xb, WvT, Vf, T_LEN, NK, HID);

    // 3) RoPE (Q,K) + V convert, all to bf16
    {
        int n = T_LEN * NQ;
        k_rope_bf16<<<(n + 255) / 256, 256, 0, stream>>>(Qf, cosb, sinb, Qb, HQ);
    }
    {
        int n = T_LEN * NK;
        k_rope_bf16<<<(n + 255) / 256, 256, 0, stream>>>(Kf, cosb, sinb, Kb, HKV);
        k_f32_to_bf16<<<(n + 255) / 256, 256, 0, stream>>>(Vf, Vb, n);
    }

    // 4) causal GQA flash attention (WMMA)
    k_flash_attn<<<dim3(T_LEN / BR, HQ), 256, 0, stream>>>(Qb, Kb, Vb, attnF);

    // 5) output projection (WMMA NT GEMM) -> d_out (f32)
    {
        int n = T_LEN * NQ;
        k_f32_to_bf16<<<(n + 255) / 256, 256, 0, stream>>>(attnF, attnB, n);
    }
    k_gemm_nt_bf16<<<dim3(HID / GN, T_LEN / GM), 256, 0, stream>>>(attnB, WoT, out, T_LEN, HID, HID);
}
